// DifferentiableStack_49417893707904
// MI455X (gfx1250) — matrix operand order repaired and checked by reading, not compiled, not run on hardware
//
#include <hip/hip_runtime.h>
#include <hip/hip_bf16.h>

// ---------------------------------------------------------------------------
// DifferentiableStack on MI455X (gfx1250, wave32, WMMA)
//
// Factored algorithm (exact in real arithmetic):
//   ptr recurrence (sequential, tiny)  -> ptrs, ptr_up, push, L (log-decay)
//   G = P @ U^T            (WMMA GEMM, 512x512x256)
//   A[t,s] = push_s * exp(L_t - L_s) * G[t,s] * [s<=t]      (epilogue, fp16)
//   reads  = A @ V + diag(exp(L)) * (P @ mem0)              (WMMA GEMM)
// ---------------------------------------------------------------------------

typedef _Float16 half_t;
typedef __attribute__((ext_vector_type(16))) _Float16 v16h;
typedef __attribute__((ext_vector_type(8)))  _Float16 v8h;
typedef __attribute__((ext_vector_type(8)))  float    v8f;

#define T_STEPS 512
#define STACK   256
#define DDIM    1024
#define EPS_F   1e-6f

// ---------------------------------------------------------------------------
// Kernel 1: sequential pointer recurrence. One workgroup, 256 threads (8 waves),
// all state in LDS. Emits: ptrs (float, to d_out tail), P/U fp16 matrices,
// push[t], and L[t] = sum_{r<=t} log(1-push_r).
// ---------------------------------------------------------------------------
__global__ __launch_bounds__(STACK) void stack_ptr_recurrence(
    const float* __restrict__ controls,
    const float* __restrict__ ptr0,
    float*  __restrict__ ptrs_out,   // [T, STACK] (second output region)
    half_t* __restrict__ P_h,        // [T, STACK] normalized ptr_t
    half_t* __restrict__ U_h,        // [T, STACK] ptr_up_t = roll(ptr_{t-1}, 1)
    float*  __restrict__ push_arr,   // [T]
    float*  __restrict__ Lsum_arr)   // [T]
{
    __shared__ float cur[STACK];
    __shared__ float red[STACK];
    const int i = threadIdx.x;
    cur[i] = ptr0[i];
    float Lrun = 0.0f;  // redundantly computed by every lane (uniform)
    __syncthreads();

    for (int t = 0; t < T_STEPS; ++t) {
        const float push = controls[t * 3 + 0];
        const float pop  = controls[t * 3 + 1];
        const float noop = controls[t * 3 + 2];

        const float up   = cur[(i + STACK - 1) & (STACK - 1)]; // roll(+1)
        const float down = cur[(i + 1) & (STACK - 1)];         // roll(-1)
        const float np   = push * up + pop * down + noop * cur[i];

        red[i] = np;
        __syncthreads();
        #pragma unroll
        for (int off = STACK / 2; off > 0; off >>= 1) {
            if (i < off) red[i] += red[i + off];
            __syncthreads();
        }
        const float newp = np / (red[0] + EPS_F);

        ptrs_out[t * STACK + i] = newp;
        P_h[t * STACK + i] = (half_t)newp;
        U_h[t * STACK + i] = (half_t)up;

        Lrun += logf(fmaxf(1.0f - push, 1e-30f));
        if (i == 0) { push_arr[t] = push; Lsum_arr[t] = Lrun; }

        __syncthreads();          // red[0] consumed by all before reuse
        cur[i] = newp;
        __syncthreads();
    }
}

// ---------------------------------------------------------------------------
// Kernel 2a/2b: fp32 -> fp16 transposed operand staging so that every WMMA
// B-operand load is K-contiguous (two b128 loads per lane).
// ---------------------------------------------------------------------------
__global__ void convert_values_t(const float* __restrict__ values,
                                 half_t* __restrict__ Vt_h) // [DDIM, T]
{
    int idx = blockIdx.x * blockDim.x + threadIdx.x;
    if (idx >= T_STEPS * DDIM) return;
    int t = idx / DDIM, d = idx - t * DDIM;
    Vt_h[d * T_STEPS + t] = (half_t)values[idx];
}

__global__ void convert_mem0_t(const float* __restrict__ mem0,
                               half_t* __restrict__ M0t_h) // [DDIM, STACK]
{
    int idx = blockIdx.x * blockDim.x + threadIdx.x;
    if (idx >= STACK * DDIM) return;
    int k = idx / DDIM, d = idx - k * DDIM;
    M0t_h[d * STACK + k] = (half_t)mem0[idx];
}

// Build the A-operand v16h per ISA 7.12.2 (16-bit A 16x32): lanes 0-15 hold
// K = {0..7,16..23}, lanes 16-31 hold K = {8..15,24..31} for row = lane%16.
__device__ __forceinline__ v16h load_wmma_a(const half_t* row, int kb, int lane)
{
    const int koff = (lane < 16) ? 0 : 8;
    v8h lo = *(const v8h*)(row + kb + koff);
    v8h hi = *(const v8h*)(row + kb + 16 + koff);
    v16h a;
    #pragma unroll
    for (int j = 0; j < 8; ++j) { a[j] = lo[j]; a[j + 8] = hi[j]; }
    return a;
}

// B-operand (16-bit B 32x16): lane holds column = lane%16; lanes 0-15 hold
// K = {0..7,16..23}, lanes 16-31 hold K = {8..15,24..31} (matches A).
__device__ __forceinline__ v16h load_wmma_b(const half_t* col, int kb, int lane)
{
    const int koff = (lane < 16) ? 0 : 8;
    v8h lo = *(const v8h*)(col + kb + koff);
    v8h hi = *(const v8h*)(col + kb + 16 + koff);
    v16h b;
    #pragma unroll
    for (int j = 0; j < 8; ++j) { b[j] = lo[j]; b[j + 8] = hi[j]; }
    return b;
}

// ---------------------------------------------------------------------------
// Kernel 3: Gram GEMM G = P @ U^T with triangular exp-decay epilogue -> A_h.
// One wave per 16x16 tile; strictly-upper tiles write zeros and skip the MACs.
// ---------------------------------------------------------------------------
__global__ __launch_bounds__(32) void gram_scale_kernel(
    const half_t* __restrict__ P_h,
    const half_t* __restrict__ U_h,
    const float*  __restrict__ push_arr,
    const float*  __restrict__ Lsum_arr,
    half_t* __restrict__ A_h)            // [T, T] fp16
{
    const int lane = threadIdx.x;
    const int s0 = blockIdx.x * 16;   // column tile (source step s)
    const int t0 = blockIdx.y * 16;   // row tile (read step t)

    if (blockIdx.x > blockIdx.y) {    // strictly upper-triangular: all zero
        #pragma unroll
        for (int r = 0; r < 8; ++r) {
            int idx = lane * 8 + r;   // 256 elements / 32 lanes
            A_h[(t0 + (idx >> 4)) * T_STEPS + s0 + (idx & 15)] = (half_t)0.0f;
        }
        return;
    }

    v8f c = {};
    const half_t* arow = P_h + (t0 + (lane & 15)) * STACK;
    const half_t* bcol = U_h + (s0 + (lane & 15)) * STACK;
    #pragma unroll
    for (int kb = 0; kb < STACK; kb += 32) {
        __builtin_prefetch(arow + kb + 32, 0, 0);
        __builtin_prefetch(bcol + kb + 32, 0, 0);
        v16h a = load_wmma_a(arow, kb, lane);
        v16h b = load_wmma_b(bcol, kb, lane);
        c = __builtin_amdgcn_wmma_f32_16x16x32_f16(false, a, false, b,
                                                   (short)0, c, false, false);
    }

    // Epilogue: C layout (ISA 7.12.2) — VGPR v holds M = v + (lane<16 ? 0 : 8),
    // N = lane % 16.
    const int n = lane & 15;
    const int mbase = (lane < 16) ? 0 : 8;
    const int s = s0 + n;
    const float pus = push_arr[s];
    const float Ls  = Lsum_arr[s];
    #pragma unroll
    for (int v = 0; v < 8; ++v) {
        const int t = t0 + mbase + v;
        float val = (s <= t) ? pus * __expf(Lsum_arr[t] - Ls) * c[v] : 0.0f;
        A_h[t * T_STEPS + s] = (half_t)val;
    }
}

// ---------------------------------------------------------------------------
// Kernel 4: reads = A @ V + diag(exp(L)) * (P @ mem0).
// One wave per 16x16 output tile. A is lower-triangular so the main K loop
// stops at the diagonal tile (halves the work on average).
// ---------------------------------------------------------------------------
__global__ __launch_bounds__(32) void read_gemm_kernel(
    const half_t* __restrict__ A_h,    // [T, T]
    const half_t* __restrict__ Vt_h,   // [DDIM, T]   (values^T)
    const half_t* __restrict__ P_h,    // [T, STACK]
    const half_t* __restrict__ M0t_h,  // [DDIM, STACK] (mem0^T)
    const float*  __restrict__ Lsum_arr,
    float* __restrict__ reads)         // [T, DDIM]
{
    const int lane = threadIdx.x;
    const int d0 = blockIdx.x * 16;
    const int t0 = blockIdx.y * 16;

    const half_t* arow  = A_h  + (t0 + (lane & 15)) * T_STEPS;
    const half_t* bcol  = Vt_h + (d0 + (lane & 15)) * T_STEPS;
    const half_t* prow  = P_h  + (t0 + (lane & 15)) * STACK;
    const half_t* m0col = M0t_h + (d0 + (lane & 15)) * STACK;

    // Main term: only K-tiles up to (and including) the diagonal contribute.
    v8f c = {};
    const int kmax = t0 + 16;
    for (int kb = 0; kb < kmax; kb += 32) {
        __builtin_prefetch(arow + kb + 32, 0, 0);
        __builtin_prefetch(bcol + kb + 32, 0, 0);
        v16h a = load_wmma_a(arow, kb, lane);
        v16h b = load_wmma_b(bcol, kb, lane);
        c = __builtin_amdgcn_wmma_f32_16x16x32_f16(false, a, false, b,
                                                   (short)0, c, false, false);
    }

    // mem0 term: c2 = (P @ mem0) tile, scaled per-row by exp(L_t) at the end.
    v8f c2 = {};
    #pragma unroll
    for (int kb = 0; kb < STACK; kb += 32) {
        v16h a = load_wmma_a(prow, kb, lane);
        v16h b = load_wmma_b(m0col, kb, lane);
        c2 = __builtin_amdgcn_wmma_f32_16x16x32_f16(false, a, false, b,
                                                    (short)0, c2, false, false);
    }

    const int n = lane & 15;
    const int mbase = (lane < 16) ? 0 : 8;
    #pragma unroll
    for (int v = 0; v < 8; ++v) {
        const int t = t0 + mbase + v;
        reads[t * DDIM + d0 + n] = c[v] + __expf(Lsum_arr[t]) * c2[v];
    }
}

// ---------------------------------------------------------------------------
// Launcher. Inputs: values [T*D], controls [T*3], mem0 [SIZE*D], ptr0 [SIZE].
// Output: reads [T*D] then ptrs [T*SIZE], concatenated flat (float32).
// Workspace layout (bytes, ~2.6 MB total):
//   P_h    @ 0        (512*256*2)
//   U_h    @ 262144   (512*256*2)
//   A_h    @ 524288   (512*512*2)
//   Vt_h   @ 1048576  (1024*512*2)
//   M0t_h  @ 2097152  (1024*256*2)
//   push   @ 2621440  (512*4)
//   Lsum   @ 2623488  (512*4)
// ---------------------------------------------------------------------------
extern "C" void kernel_launch(void* const* d_in, const int* in_sizes, int n_in,
                              void* d_out, int out_size, void* d_ws, size_t ws_size,
                              hipStream_t stream)
{
    const float* values   = (const float*)d_in[0];
    const float* controls = (const float*)d_in[1];
    const float* mem0     = (const float*)d_in[2];
    const float* ptr0     = (const float*)d_in[3];

    float* reads_out = (float*)d_out;                      // [T, DDIM]
    float* ptrs_out  = (float*)d_out + T_STEPS * DDIM;     // [T, STACK]

    char* ws = (char*)d_ws;
    half_t* P_h     = (half_t*)(ws + 0);
    half_t* U_h     = (half_t*)(ws + 262144);
    half_t* A_h     = (half_t*)(ws + 524288);
    half_t* Vt_h    = (half_t*)(ws + 1048576);
    half_t* M0t_h   = (half_t*)(ws + 2097152);
    float*  push_a  = (float*) (ws + 2621440);
    float*  Lsum_a  = (float*) (ws + 2623488);

    // Operand staging (independent of the recurrence; stream-ordered anyway).
    convert_values_t<<<(T_STEPS * DDIM + 255) / 256, 256, 0, stream>>>(values, Vt_h);
    convert_mem0_t<<<(STACK * DDIM + 255) / 256, 256, 0, stream>>>(mem0, M0t_h);

    // Sequential pointer recurrence (single WGP, LDS-resident).
    stack_ptr_recurrence<<<1, STACK, 0, stream>>>(controls, ptr0, ptrs_out,
                                                  P_h, U_h, push_a, Lsum_a);

    // Gram GEMM + triangular decay epilogue -> A (fp16).
    gram_scale_kernel<<<dim3(T_STEPS / 16, T_STEPS / 16), 32, 0, stream>>>(
        P_h, U_h, push_a, Lsum_a, A_h);

    // reads = A @ V + diag(exp(L)) * (P @ mem0).
    read_gemm_kernel<<<dim3(DDIM / 16, T_STEPS / 16), 32, 0, stream>>>(
        A_h, Vt_h, P_h, M0t_h, Lsum_a, reads_out);
}
